// MultiheadCosformerAttention__48241072668682
// MI455X (gfx1250) — matrix operand order repaired and checked by reading, not compiled, not run on hardware
//
#include <hip/hip_runtime.h>
#include <hip/hip_bf16.h>

// ---------------------------------------------------------------------------
// cosFormer multihead linear attention for MI455X (gfx1250), wave32 + WMMA.
// Pipeline: f16 convert -> QKV WMMA GEMMs (async-LDS double buffered)
//          -> per-head KV-state WMMA -> ksum/z -> attention-apply WMMA
//          -> output-projection WMMA (f32 out).
// ---------------------------------------------------------------------------

typedef __attribute__((ext_vector_type(16))) _Float16 v16h;
typedef __attribute__((ext_vector_type(8)))  _Float16 v8h;
typedef __attribute__((ext_vector_type(8)))  float    v8f;

#define BSZ   4
#define LSEQ  4096
#define EDIM  512
#define HEADS 8
#define DHEAD 64
#define MROWS (BSZ * LSEQ)      // 16384
#define NSEG  8                 // L-split for KV-state kernel
#define LPSEG (LSEQ / NSEG)     // 512

union pack2 {
  _Float16 h[2];
  uint32_t u;
};

__device__ __forceinline__ float ang_cos(int l) {
  const float c = 3.14159265358979323846f * 0.5f / (float)LSEQ;
  return __cosf(c * (float)(l + 1));
}
__device__ __forceinline__ float ang_sin(int l) {
  const float c = 3.14159265358979323846f * 0.5f / (float)LSEQ;
  return __sinf(c * (float)(l + 1));
}

__device__ __forceinline__ v16h cat8(v8h lo, v8h hi) {
  return __builtin_shufflevector(lo, hi, 0, 1, 2, 3, 4, 5, 6, 7,
                                         8, 9, 10, 11, 12, 13, 14, 15);
}

// A-fragment (16x32, f16) from row-major src (ld in halfs).
// Lane layout: row = row0 + lane%16; K halves: (lane/16)*8 + {0..7, 16..23}.
__device__ __forceinline__ v16h load_frag_a(const _Float16* src, int ld,
                                            int row0, int k0, int lane) {
  int m  = lane & 15;
  int kb = (lane >> 4) * 8;
  const _Float16* p = src + (size_t)(row0 + m) * ld + k0 + kb;
  v8h lo = *(const v8h*)(p);
  v8h hi = *(const v8h*)(p + 16);
  return cat8(lo, hi);
}

// B-fragment (32x16, f16) where B[k][n] = src[n][k] (src row-major N x K).
// Lane layout: col = n0 + lane%16; K halves: (lane/16)*8 + {0..7, 16..23}.
__device__ __forceinline__ v16h load_frag_bT(const _Float16* src, int ld,
                                             int n0, int k0, int lane) {
  int n  = lane & 15;
  int kb = (lane >> 4) * 8;
  const _Float16* p = src + (size_t)(n0 + n) * ld + k0 + kb;
  v8h lo = *(const v8h*)(p);
  v8h hi = *(const v8h*)(p + 16);
  return cat8(lo, hi);
}

#define WMMA_F16(a, b, c) \
  __builtin_amdgcn_wmma_f32_16x16x32_f16(false, (a), false, (b), (short)0, (c), false, false)

// Async global -> LDS 16-byte copy (CDNA5 GLOBAL_LOAD_ASYNC_TO_LDS_B128).
// lds_off: wave-relative LDS byte offset (low 32 bits of generic pointer).
__device__ __forceinline__ void async_ld16(uint32_t lds_off, const void* gptr) {
  asm volatile("global_load_async_to_lds_b128 %0, %1, off"
               :: "v"(lds_off), "v"((unsigned long long)(uintptr_t)gptr)
               : "memory");
}

// ---------------------------------------------------------------------------
// f32 -> f16 conversion
// ---------------------------------------------------------------------------
__global__ void cvt_f32_f16(const float* __restrict__ src,
                            _Float16* __restrict__ dst, int n) {
  int i = blockIdx.x * 256 + threadIdx.x;
  if (i < n) dst[i] = (_Float16)src[i];
}

// ---------------------------------------------------------------------------
// C[m,n] = act( sum_k A[m,k] * W[n,k] + bias[n] ).  A: Mx512 f16, W: 512x512 f16.
// Block: 256 thr = 8 waves as 4(m) x 2(n) -> 128x128 tile. Wave tile: 32x64.
// A/B tiles double-buffered in LDS via async copies tracked with ASYNCcnt.
// ---------------------------------------------------------------------------
template <bool RELU, bool OUT16>
__global__ void gemm_xwT(const _Float16* __restrict__ A,
                         const _Float16* __restrict__ W,
                         const float* __restrict__ bias,
                         void* __restrict__ Cout, int Ndim, int Kdim) {
  __shared__ _Float16 Abuf[2][128][40];  // 128 rows x 32 k (+8 pad)
  __shared__ _Float16 Bbuf[2][128][40];

  int t    = threadIdx.x;
  int lane = t & 31;
  int wv   = t >> 5;
  int wm   = wv & 3, wn = wv >> 2;
  int rowB = blockIdx.x * 128;
  int colB = blockIdx.y * 128;

  auto issue = [&](int buf, int k0) {
    // 512 x 16B chunks per tile; 2 A + 2 B chunks per thread.
#pragma unroll
    for (int u = 0; u < 2; ++u) {
      int idx = t + 256 * u;
      int r = idx >> 1;
      int c = (idx & 1) * 8;  // half offset within the 32-wide k slab
      async_ld16((uint32_t)(uintptr_t)&Abuf[buf][r][c],
                 A + (size_t)(rowB + r) * Kdim + k0 + c);
      async_ld16((uint32_t)(uintptr_t)&Bbuf[buf][r][c],
                 W + (size_t)(colB + r) * Kdim + k0 + c);
    }
  };

  issue(0, 0);

  v8f acc[2][4] = {};
  const int steps = Kdim / 32;
  for (int s = 0; s < steps; ++s) {
    int cur = s & 1;
    if (s + 1 < steps) {
      issue(cur ^ 1, (s + 1) * 32);
      // 8 outstanding; first 4 (current buffer) complete in order.
      asm volatile("s_wait_asynccnt 0x4" ::: "memory");
    } else {
      asm volatile("s_wait_asynccnt 0x0" ::: "memory");
    }
    __syncthreads();

    const _Float16* Ab = &Abuf[cur][0][0];
    const _Float16* Bb = &Bbuf[cur][0][0];
    v16h a0 = load_frag_a(Ab, 40, wm * 32, 0, lane);
    v16h a1 = load_frag_a(Ab, 40, wm * 32 + 16, 0, lane);
#pragma unroll
    for (int j = 0; j < 4; ++j) {
      v16h b = load_frag_bT(Bb, 40, wn * 64 + j * 16, 0, lane);
      acc[0][j] = WMMA_F16(a0, b, acc[0][j]);
      acc[1][j] = WMMA_F16(a1, b, acc[1][j]);
    }
    __syncthreads();  // reads done before this buffer is refilled
  }

  int row0 = rowB + wm * 32;
  int col0 = colB + wn * 64;
  int nl = lane & 15;
  int rb = (lane >> 4) * 8;
#pragma unroll
  for (int i = 0; i < 2; ++i) {
#pragma unroll
    for (int j = 0; j < 4; ++j) {
      int col = col0 + j * 16 + nl;
      float bv = bias[col];
#pragma unroll
      for (int r = 0; r < 8; ++r) {
        float v = acc[i][j][r] + bv;
        if (RELU) v = fmaxf(v, 0.0f);
        size_t idx = (size_t)(row0 + i * 16 + rb + r) * Ndim + col;
        if (OUT16) ((_Float16*)Cout)[idx] = (_Float16)v;
        else       ((float*)Cout)[idx]    = v;
      }
    }
  }
}

// ---------------------------------------------------------------------------
// KV state partials: for (b,h,seg), kv_cs[kd,vd] += sum_l w_cs(l)*k[l,kd]*v[l,vd]
// Stage k*cos, k*sin, v transposed ([dim][l]) in LDS (packed b32 writes),
// contract over L via WMMA. Waves 0..3 -> cos, waves 4..7 -> sin.
// ---------------------------------------------------------------------------
__global__ void kv_partial(const _Float16* __restrict__ Kmat,
                           const _Float16* __restrict__ Vmat,
                           float* __restrict__ kvpart) {
  __shared__ _Float16 kc[64][72];
  __shared__ _Float16 ks[64][72];
  __shared__ _Float16 vt[64][72];

  int bh = blockIdx.x, seg = blockIdx.y;
  int b = bh >> 3, h = bh & 7;
  int lane = threadIdx.x & 31, wv = threadIdx.x >> 5;
  int cs = wv >> 2, mi = wv & 3;

  v8f acc[4] = {};

  int pl = threadIdx.x >> 3;        // l-pair index 0..31
  int td = (threadIdx.x & 7) * 8;   // dim group of 8

  for (int lc = 0; lc < LPSEG; lc += 64) {
    int lg0 = seg * LPSEG + lc + pl * 2;
    int lg1 = lg0 + 1;
    float c0 = ang_cos(lg0), s0 = ang_sin(lg0);
    float c1 = ang_cos(lg1), s1 = ang_sin(lg1);
    const _Float16* kp0 = Kmat + ((size_t)b * LSEQ + lg0) * EDIM + h * 64 + td;
    const _Float16* kp1 = kp0 + EDIM;
    const _Float16* vp0 = Vmat + ((size_t)b * LSEQ + lg0) * EDIM + h * 64 + td;
    const _Float16* vp1 = vp0 + EDIM;
#pragma unroll
    for (int e = 0; e < 8; ++e) {
      float k0f = (float)kp0[e], k1f = (float)kp1[e];
      pack2 pc, ps, pv;
      pc.h[0] = (_Float16)(k0f * c0); pc.h[1] = (_Float16)(k1f * c1);
      ps.h[0] = (_Float16)(k0f * s0); ps.h[1] = (_Float16)(k1f * s1);
      pv.h[0] = vp0[e];               pv.h[1] = vp1[e];
      *(uint32_t*)&kc[td + e][pl * 2] = pc.u;
      *(uint32_t*)&ks[td + e][pl * 2] = ps.u;
      *(uint32_t*)&vt[td + e][pl * 2] = pv.u;
    }
    __syncthreads();

    const _Float16* ka = cs ? &ks[0][0] : &kc[0][0];
#pragma unroll
    for (int kk = 0; kk < 64; kk += 32) {
      v16h a = load_frag_a(ka, 72, mi * 16, kk, lane);
#pragma unroll
      for (int j = 0; j < 4; ++j) {
        v16h bfr = load_frag_bT(&vt[0][0], 72, j * 16, kk, lane);
        acc[j] = WMMA_F16(a, bfr, acc[j]);
      }
    }
    __syncthreads();
  }

  float* outp = kvpart + ((((size_t)seg * (BSZ * HEADS) + bh) * 2 + cs) * 64) * 64;
  int vdl = lane & 15, rb = (lane >> 4) * 8;
#pragma unroll
  for (int j = 0; j < 4; ++j)
#pragma unroll
    for (int r = 0; r < 8; ++r) {
      int kd = mi * 16 + rb + r;
      int vd = j * 16 + vdl;
      outp[kd * 64 + vd] = acc[j][r];
    }
}

// ---------------------------------------------------------------------------
// ksum_cs[b,h,kd] = sum_l w_cs(l) * k[l,kd]
// ---------------------------------------------------------------------------
__global__ void ksum_kernel(const _Float16* __restrict__ Kmat,
                            float* __restrict__ ksum) {
  __shared__ float red[2][4][64];
  int bh = blockIdx.x;
  int b = bh >> 3, h = bh & 7;
  int kd  = threadIdx.x & 63;
  int grp = threadIdx.x >> 6;
  float sc = 0.0f, ss = 0.0f;
  for (int l = grp; l < LSEQ; l += 4) {
    float kf = (float)Kmat[((size_t)b * LSEQ + l) * EDIM + h * 64 + kd];
    sc += kf * ang_cos(l);
    ss += kf * ang_sin(l);
  }
  red[0][grp][kd] = sc;
  red[1][grp][kd] = ss;
  __syncthreads();
  if (threadIdx.x < 128) {
    int csel = threadIdx.x >> 6, d = threadIdx.x & 63;
    float s = red[csel][0][d] + red[csel][1][d] + red[csel][2][d] + red[csel][3][d];
    ksum[((size_t)bh * 2 + csel) * 64 + d] = s;
  }
}

// ---------------------------------------------------------------------------
// Reduce segment partials -> f16 KV states
// ---------------------------------------------------------------------------
__global__ void kv_reduce(const float* __restrict__ kvpart,
                          _Float16* __restrict__ kvh) {
  size_t i = (size_t)blockIdx.x * 256 + threadIdx.x;  // < 32*2*4096
  const size_t stride = (size_t)BSZ * HEADS * 2 * 64 * 64;
  float s = 0.0f;
#pragma unroll
  for (int seg = 0; seg < NSEG; ++seg) s += kvpart[(size_t)seg * stride + i];
  kvh[i] = (_Float16)s;
}

// ---------------------------------------------------------------------------
// attn[l,vd] = z(l) * sum_k ( qc[l,k]*kv_c[k,vd] + qs[l,k]*kv_s[k,vd] )
// Block: (lblk of 256 rows, bh). KV staged transposed in LDS (packed b32);
// z in LDS.
// ---------------------------------------------------------------------------
__global__ void attn_apply(const _Float16* __restrict__ Q,
                           const _Float16* __restrict__ kvh,
                           const float* __restrict__ ksum,
                           _Float16* __restrict__ attn) {
  __shared__ _Float16 kvt[2][64][72];  // [cs][vd][kd]
  __shared__ float zl[256];
  __shared__ float ksl[2][64];

  int bh = blockIdx.y;
  int b = bh >> 3, h = bh & 7;
  int l0 = blockIdx.x * 256;
  int t = threadIdx.x;

  {  // stage KV transposed (kd-pairs packed) + ksum
    int p   = t >> 3;         // kd pair 0..31
    int vd0 = (t & 7) * 8;    // vd group of 8
#pragma unroll
    for (int cs = 0; cs < 2; ++cs) {
      const _Float16* r0 = kvh + (((size_t)bh * 2 + cs) * 64 + 2 * p) * 64 + vd0;
      const _Float16* r1 = r0 + 64;
#pragma unroll
      for (int e = 0; e < 8; ++e) {
        pack2 w;
        w.h[0] = r0[e];
        w.h[1] = r1[e];
        *(uint32_t*)&kvt[cs][vd0 + e][2 * p] = w.u;
      }
    }
    if (t < 128) ksl[t >> 6][t & 63] = ksum[((size_t)bh * 2 + (t >> 6)) * 64 + (t & 63)];
  }
  __syncthreads();

  {  // z per row
    int l = l0 + t;
    float cl = ang_cos(l), sl = ang_sin(l);
    const _Float16* qp = Q + ((size_t)b * LSEQ + l) * EDIM + h * 64;
    float dc = 0.0f, dsn = 0.0f;
    for (int e = 0; e < 64; ++e) {
      float qv = (float)qp[e];
      dc  += qv * ksl[0][e];
      dsn += qv * ksl[1][e];
    }
    zl[t] = 1.0f / fmaxf(cl * dc + sl * dsn, 1e-6f);
  }
  __syncthreads();

  int lane = t & 31, wv = t >> 5;
  int rbase = wv * 32;  // 32 rows per wave
  v8f acc[2][4] = {};

  for (int kk = 0; kk < 64; kk += 32) {
#pragma unroll
    for (int i = 0; i < 2; ++i) {
      int m  = lane & 15;
      int kb = (lane >> 4) * 8;
      int lr = l0 + rbase + i * 16 + m;
      const _Float16* p = Q + ((size_t)b * LSEQ + lr) * EDIM + h * 64 + kk + kb;
      v8h lo = *(const v8h*)(p);
      v8h hi = *(const v8h*)(p + 16);
      v16h qa = cat8(lo, hi);
      float c2 = ang_cos(lr), s2 = ang_sin(lr);
      v16h qc, qs;
#pragma unroll
      for (int e = 0; e < 16; ++e) {
        float f = (float)qa[e];
        qc[e] = (_Float16)(f * c2);
        qs[e] = (_Float16)(f * s2);
      }
#pragma unroll
      for (int j = 0; j < 4; ++j) {
        v16h bc = load_frag_bT(&kvt[0][0][0], 72, j * 16, kk, lane);
        v16h bs = load_frag_bT(&kvt[1][0][0], 72, j * 16, kk, lane);
        acc[i][j] = WMMA_F16(qc, bc, acc[i][j]);
        acc[i][j] = WMMA_F16(qs, bs, acc[i][j]);
      }
    }
  }

  int vdl = lane & 15, rb8 = (lane >> 4) * 8;
#pragma unroll
  for (int i = 0; i < 2; ++i)
#pragma unroll
    for (int j = 0; j < 4; ++j)
#pragma unroll
      for (int r = 0; r < 8; ++r) {
        int lloc = rbase + i * 16 + rb8 + r;
        float v = acc[i][j][r] * zl[lloc];
        attn[((size_t)b * LSEQ + l0 + lloc) * EDIM + h * 64 + j * 16 + vdl] =
            (_Float16)v;
      }
}

// ---------------------------------------------------------------------------
// Host launcher
// ---------------------------------------------------------------------------
extern "C" void kernel_launch(void* const* d_in, const int* in_sizes, int n_in,
                              void* d_out, int out_size, void* d_ws, size_t ws_size,
                              hipStream_t stream) {
  const float* x  = (const float*)d_in[0];
  const float* Wq = (const float*)d_in[1];
  const float* bq = (const float*)d_in[2];
  const float* Wk = (const float*)d_in[3];
  const float* bk = (const float*)d_in[4];
  const float* Wv = (const float*)d_in[5];
  const float* bv = (const float*)d_in[6];
  const float* Wo = (const float*)d_in[7];
  const float* bo = (const float*)d_in[8];
  float* out = (float*)d_out;

  // workspace carve-out (256B aligned)
  char* ws = (char*)d_ws;
  size_t off = 0;
  auto alloc = [&](size_t bytes) -> char* {
    char* p = ws + off;
    off += (bytes + 255) & ~(size_t)255;
    return p;
  };
  const size_t nXE = (size_t)MROWS * EDIM;
  const size_t nWW = (size_t)EDIM * EDIM;
  const size_t nKV = (size_t)BSZ * HEADS * 2 * 64 * 64;

  _Float16* xh   = (_Float16*)alloc(nXE * 2);
  _Float16* wqh  = (_Float16*)alloc(nWW * 2);
  _Float16* wkh  = (_Float16*)alloc(nWW * 2);
  _Float16* wvh  = (_Float16*)alloc(nWW * 2);
  _Float16* woh  = (_Float16*)alloc(nWW * 2);
  _Float16* qh   = (_Float16*)alloc(nXE * 2);
  _Float16* kh   = (_Float16*)alloc(nXE * 2);
  _Float16* vh   = (_Float16*)alloc(nXE * 2);
  float*    kvp  = (float*)   alloc((size_t)NSEG * nKV * 4);
  _Float16* kvh  = (_Float16*)alloc(nKV * 2);
  float*    ksum = (float*)   alloc((size_t)BSZ * HEADS * 2 * 64 * 4);
  _Float16* ath  = (_Float16*)alloc(nXE * 2);

  // 1) f16 conversions
  cvt_f32_f16<<<(int)((nXE + 255) / 256), 256, 0, stream>>>(x, xh, (int)nXE);
  cvt_f32_f16<<<(int)((nWW + 255) / 256), 256, 0, stream>>>(Wq, wqh, (int)nWW);
  cvt_f32_f16<<<(int)((nWW + 255) / 256), 256, 0, stream>>>(Wk, wkh, (int)nWW);
  cvt_f32_f16<<<(int)((nWW + 255) / 256), 256, 0, stream>>>(Wv, wvh, (int)nWW);
  cvt_f32_f16<<<(int)((nWW + 255) / 256), 256, 0, stream>>>(Wo, woh, (int)nWW);

  // 2) projections: q = relu(xWq^T+bq), k = relu(xWk^T+bk), v = xWv^T+bv
  dim3 ggrid(MROWS / 128, EDIM / 128);
  gemm_xwT<true,  true><<<ggrid, 256, 0, stream>>>(xh, wqh, bq, qh, EDIM, EDIM);
  gemm_xwT<true,  true><<<ggrid, 256, 0, stream>>>(xh, wkh, bk, kh, EDIM, EDIM);
  gemm_xwT<false, true><<<ggrid, 256, 0, stream>>>(xh, wvh, bv, vh, EDIM, EDIM);

  // 3) KV states (segmented) + reduce, 4) k-sums
  kv_partial<<<dim3(BSZ * HEADS, NSEG), 256, 0, stream>>>(kh, vh, kvp);
  ksum_kernel<<<BSZ * HEADS, 256, 0, stream>>>(kh, ksum);
  kv_reduce<<<(int)(nKV / 256), 256, 0, stream>>>(kvp, kvh);

  // 5) attention apply (z-normalized)
  attn_apply<<<dim3(LSEQ / 256, BSZ * HEADS), 256, 0, stream>>>(qh, kvh, ksum, ath);

  // 6) output projection -> f32 out
  gemm_xwT<false, false><<<ggrid, 256, 0, stream>>>(ath, woh, bo, out, EDIM, EDIM);

  (void)in_sizes; (void)n_in; (void)out_size; (void)ws_size;
}
